// TopKMoE_86079734546615
// MI455X (gfx1250) — compile-verified, matrix operand order checked by reading, not run on hardware
//
#include <hip/hip_runtime.h>
#include <math.h>

// ---------------- problem constants ----------------
#define E_        8
#define KSEL      2
#define IN_       512
#define HID_      1024
#define OUT_      512
#define BTOK      16384
#define SLOTS_PAD 32896            // 2*BTOK + 8*16, multiple of 16
#define ROW_TILES (SLOTS_PAD / 16) // 2056
#define NT_       4                // 16x64 output tile per wave

// ---------------- vector types for WMMA ----------------
typedef __attribute__((ext_vector_type(16))) __bf16         v16bf;
typedef __attribute__((ext_vector_type(8)))  float          v8f;
typedef __attribute__((ext_vector_type(8)))  unsigned short v8u16;

union Frag16 {
    v16bf bf;
    v8u16 u[2];
};

// ---------------- helpers ----------------
__device__ __forceinline__ unsigned short f2bf(float f) {
    unsigned int u = __float_as_uint(f);
    u += 0x7FFFu + ((u >> 16) & 1u);      // round-to-nearest-even
    return (unsigned short)(u >> 16);
}

__device__ __forceinline__ float gelu_exact(float x) {
    return 0.5f * x * (1.0f + erff(x * 0.70710678118654752440f));
}

// ---------------- kernel 1: init all mutable state ----------------
__global__ void k_init(float* __restrict__ out, int* __restrict__ perm,
                       float* __restrict__ slotw, int* __restrict__ counts,
                       int* __restrict__ cursors, int* __restrict__ poff,
                       unsigned short* __restrict__ zrow) {
    long long tid    = (long long)blockIdx.x * blockDim.x + threadIdx.x;
    long long stride = (long long)gridDim.x * blockDim.x;
    const long long OUTN = (long long)BTOK * OUT_ + 1;
    for (long long i = tid; i < OUTN; i += stride) out[i] = 0.0f;
    for (long long i = tid; i < SLOTS_PAD; i += stride) { perm[i] = -1; slotw[i] = 0.0f; }
    if (tid < IN_)    { zrow[tid] = 0; }
    if (tid < E_)     { counts[tid] = 0; cursors[tid] = 0; }
    if (tid < E_ + 1) { poff[tid] = 0; }
}

// ---------------- kernels 2-4: fp32 -> bf16 conversions ----------------
__global__ void k_cvt_x(const float* __restrict__ x, unsigned short* __restrict__ x16) {
    long long i = (long long)blockIdx.x * blockDim.x + threadIdx.x;
    if (i < (long long)BTOK * IN_) x16[i] = f2bf(x[i]);
}

// w1t[e][h][i] = w1[e][i][h]
__global__ void k_cvt_w1t(const float* __restrict__ w1, unsigned short* __restrict__ w1t) {
    long long t = (long long)blockIdx.x * blockDim.x + threadIdx.x;
    if (t >= (long long)E_ * HID_ * IN_) return;
    int       i = (int)(t % IN_);
    long long r = t / IN_;
    int       h = (int)(r % HID_);
    int       e = (int)(r / HID_);
    w1t[t] = f2bf(w1[((long long)e * IN_ + i) * HID_ + h]);
}

// w2t[e][o][h] = w2[e][h][o]
__global__ void k_cvt_w2t(const float* __restrict__ w2, unsigned short* __restrict__ w2t) {
    long long t = (long long)blockIdx.x * blockDim.x + threadIdx.x;
    if (t >= (long long)E_ * OUT_ * HID_) return;
    int       h = (int)(t % HID_);
    long long r = t / HID_;
    int       o = (int)(r % OUT_);
    int       e = (int)(r / OUT_);
    w2t[t] = f2bf(w2[((long long)e * HID_ + h) * OUT_ + o]);
}

// ---------------- kernel 5: gating (one wave32 per token, f32) ----------------
__global__ void k_gate(const float* __restrict__ x, const float* __restrict__ gw,
                       const float* __restrict__ gb, int* __restrict__ tki,
                       float* __restrict__ tkw, int* __restrict__ counts) {
    int wid   = threadIdx.x >> 5;
    int lane  = threadIdx.x & 31;
    int token = blockIdx.x * (blockDim.x >> 5) + wid;
    if (token >= BTOK) return;

    float acc[E_];
#pragma unroll
    for (int e = 0; e < E_; ++e) acc[e] = 0.0f;

    const float* xr = x + (long long)token * IN_;
    for (int d = lane; d < IN_; d += 32) {
        float xv = xr[d];
#pragma unroll
        for (int e = 0; e < E_; ++e) acc[e] += xv * gw[d * E_ + e];
    }
#pragma unroll
    for (int e = 0; e < E_; ++e) {
#pragma unroll
        for (int off = 16; off > 0; off >>= 1) acc[e] += __shfl_xor(acc[e], off, 32);
    }

    if (lane == 0) {
        float s[E_];
#pragma unroll
        for (int e = 0; e < E_; ++e) s[e] = acc[e] + gb[e];
        // top-2 with jax tie semantics (first index wins, strict >)
        int   bi = 0;  float bv = s[0];
#pragma unroll
        for (int e = 1; e < E_; ++e) if (s[e] > bv) { bv = s[e]; bi = e; }
        int   si = (bi == 0) ? 1 : 0;  float sv = s[si];
#pragma unroll
        for (int e = 0; e < E_; ++e) if (e != bi && s[e] > sv) { sv = s[e]; si = e; }
        // softmax over {bv, sv}; bv >= sv
        float e1  = __expf(sv - bv);
        float inv = 1.0f / (1.0f + e1);
        tki[token * 2 + 0] = bi;  tkw[token * 2 + 0] = inv;
        tki[token * 2 + 1] = si;  tkw[token * 2 + 1] = e1 * inv;
        atomicAdd(&counts[bi], 1);
        atomicAdd(&counts[si], 1);
    }
}

// ---------------- kernel 6: segment offsets + aux loss ----------------
__global__ void k_offsets(const int* __restrict__ counts, int* __restrict__ poff,
                          float* __restrict__ aux_out) {
    if (blockIdx.x == 0 && threadIdx.x == 0) {
        int   off = 0;
        float aux = 0.0f;
        poff[0] = 0;
        for (int e = 0; e < E_; ++e) {
            int c = counts[e];
            off += (c + 15) & ~15;           // 16-align each expert segment
            poff[e + 1] = off;
            float frac = (float)c / (float)(BTOK * KSEL);
            float d    = frac - (1.0f / (float)E_);
            aux += d * d;
        }
        aux_out[0] = aux / (float)E_;
    }
}

// ---------------- kernel 7: scatter (token,k) -> slots ----------------
__global__ void k_scatter(const int* __restrict__ tki, const float* __restrict__ tkw,
                          const int* __restrict__ poff, int* __restrict__ cursors,
                          int* __restrict__ perm, float* __restrict__ slotw) {
    int i = blockIdx.x * blockDim.x + threadIdx.x;
    if (i >= BTOK * KSEL) return;
    int e    = tki[i];
    int pos  = atomicAdd(&cursors[e], 1);
    int slot = poff[e] + pos;
    perm[slot]  = i >> 1;
    slotw[slot] = tkw[i];
}

// ---------------- kernel 8: GEMM1 (x @ w1) + bias + GELU -> h (bf16) --------
// One wave computes a 16x64 output tile: one shared A fragment, 4 accumulators.
__global__ void __launch_bounds__(32)
k_gemm1(const unsigned short* __restrict__ x16, const unsigned short* __restrict__ w1t,
        const float* __restrict__ b1, const int* __restrict__ perm,
        const int* __restrict__ poff, unsigned short* __restrict__ h16,
        const unsigned short* __restrict__ zrow) {
    int row0  = blockIdx.x * 16;
    int total = poff[E_];
    if (row0 >= total) return;
    int e = 0;
    while (e < E_ - 1 && poff[e + 1] <= row0) ++e;

    int col0 = blockIdx.y * (16 * NT_);
    int lane = threadIdx.x;
    int mn   = lane & 15;      // A row index / B col index / D col index
    int half = lane >> 4;      // K sub-chunk select per ISA 16-bit layout

    int token = perm[row0 + mn];
    // padding lanes read a zeroed dummy row: no divergent branch in the K-loop
    const unsigned short* arow  = (token >= 0) ? (x16 + (long long)token * IN_) : zrow;
    const unsigned short* bbase = w1t + ((long long)e * HID_ + col0 + mn) * IN_;

    v8f acc[NT_];
#pragma unroll
    for (int t = 0; t < NT_; ++t) acc[t] = (v8f){};

    for (int kb = 0; kb < IN_; kb += 32) {
        Frag16 a;
        a.u[0] = *(const v8u16*)(arow + kb + half * 8);
        a.u[1] = *(const v8u16*)(arow + kb + 16 + half * 8);
        if (kb + 32 < IN_) __builtin_prefetch(arow + kb + 32, 0, 1);
#pragma unroll
        for (int t = 0; t < NT_; ++t) {
            const unsigned short* bcol = bbase + (long long)t * 16 * IN_;
            Frag16 b;
            b.u[0] = *(const v8u16*)(bcol + kb + half * 8);
            b.u[1] = *(const v8u16*)(bcol + kb + 16 + half * 8);
            acc[t] = __builtin_amdgcn_wmma_f32_16x16x32_bf16(false, a.bf, false, b.bf,
                                                             (short)0, acc[t], false, false);
        }
    }

#pragma unroll
    for (int t = 0; t < NT_; ++t) {
        int   col  = col0 + t * 16 + mn;
        float bias = b1[e * HID_ + col];
#pragma unroll
        for (int r = 0; r < 8; ++r) {
            int   row = half * 8 + r;
            float hv  = gelu_exact(acc[t][r] + bias);
            h16[(long long)(row0 + row) * HID_ + col] = f2bf(hv);
        }
    }
}

// ---------------- kernel 9: GEMM2 (h @ w2) + bias, weighted scatter-add -----
__global__ void __launch_bounds__(32)
k_gemm2(const unsigned short* __restrict__ h16, const unsigned short* __restrict__ w2t,
        const float* __restrict__ b2, const int* __restrict__ perm,
        const float* __restrict__ slotw, const int* __restrict__ poff,
        float* __restrict__ out) {
    int row0  = blockIdx.x * 16;
    int total = poff[E_];
    if (row0 >= total) return;
    int e = 0;
    while (e < E_ - 1 && poff[e + 1] <= row0) ++e;

    int col0 = blockIdx.y * (16 * NT_);
    int lane = threadIdx.x;
    int mn   = lane & 15;
    int half = lane >> 4;

    const unsigned short* arow  = h16 + (long long)(row0 + mn) * HID_;
    const unsigned short* bbase = w2t + ((long long)e * OUT_ + col0 + mn) * HID_;

    v8f acc[NT_];
#pragma unroll
    for (int t = 0; t < NT_; ++t) acc[t] = (v8f){};

    for (int kb = 0; kb < HID_; kb += 32) {
        Frag16 a;
        a.u[0] = *(const v8u16*)(arow + kb + half * 8);
        a.u[1] = *(const v8u16*)(arow + kb + 16 + half * 8);
        if (kb + 32 < HID_) __builtin_prefetch(arow + kb + 32, 0, 1);
#pragma unroll
        for (int t = 0; t < NT_; ++t) {
            const unsigned short* bcol = bbase + (long long)t * 16 * HID_;
            Frag16 b;
            b.u[0] = *(const v8u16*)(bcol + kb + half * 8);
            b.u[1] = *(const v8u16*)(bcol + kb + 16 + half * 8);
            acc[t] = __builtin_amdgcn_wmma_f32_16x16x32_bf16(false, a.bf, false, b.bf,
                                                             (short)0, acc[t], false, false);
        }
    }

#pragma unroll
    for (int t = 0; t < NT_; ++t) {
        int   col  = col0 + t * 16 + mn;
        float bias = b2[e * OUT_ + col];
#pragma unroll
        for (int r = 0; r < 8; ++r) {
            int slot  = row0 + half * 8 + r;
            int token = perm[slot];
            if (token >= 0) {
                float v = slotw[slot] * (acc[t][r] + bias);
                atomicAdd(&out[(long long)token * OUT_ + col], v);
            }
        }
    }
}

// ---------------- host launcher ----------------
extern "C" void kernel_launch(void* const* d_in, const int* in_sizes, int n_in,
                              void* d_out, int out_size, void* d_ws, size_t ws_size,
                              hipStream_t stream) {
    const float* x      = (const float*)d_in[0];
    const float* gate_w = (const float*)d_in[1];
    const float* gate_b = (const float*)d_in[2];
    const float* w1     = (const float*)d_in[3];
    const float* b1     = (const float*)d_in[4];
    const float* w2     = (const float*)d_in[5];
    const float* b2     = (const float*)d_in[6];
    float*       out    = (float*)d_out;

    // workspace carve (~102 MB)
    char*  ws  = (char*)d_ws;
    size_t off = 0;
    auto carve = [&](size_t bytes) -> void* {
        void* p = ws + off;
        off = (off + bytes + 255) & ~(size_t)255;
        return p;
    };
    unsigned short* x16   = (unsigned short*)carve((size_t)BTOK * IN_ * 2);
    unsigned short* w1t   = (unsigned short*)carve((size_t)E_ * HID_ * IN_ * 2);
    unsigned short* w2t   = (unsigned short*)carve((size_t)E_ * OUT_ * HID_ * 2);
    unsigned short* h16   = (unsigned short*)carve((size_t)SLOTS_PAD * HID_ * 2);
    float*          slotw = (float*)carve((size_t)SLOTS_PAD * 4);
    int*            perm  = (int*)carve((size_t)SLOTS_PAD * 4);
    int*            tki   = (int*)carve((size_t)BTOK * KSEL * 4);
    float*          tkw   = (float*)carve((size_t)BTOK * KSEL * 4);
    unsigned short* zrow  = (unsigned short*)carve((size_t)IN_ * 2);
    int*            counts  = (int*)carve(64);
    int*            cursors = (int*)carve(64);
    int*            poff    = (int*)carve(64);

    (void)in_sizes; (void)n_in; (void)out_size; (void)ws_size;

    k_init<<<4096, 256, 0, stream>>>(out, perm, slotw, counts, cursors, poff, zrow);
    k_cvt_x  <<<(BTOK * IN_) / 256,        256, 0, stream>>>(x,  x16);
    k_cvt_w1t<<<(E_ * HID_ * IN_) / 256,   256, 0, stream>>>(w1, w1t);
    k_cvt_w2t<<<(E_ * OUT_ * HID_) / 256,  256, 0, stream>>>(w2, w2t);
    k_gate   <<<BTOK / 8, 256, 0, stream>>>(x, gate_w, gate_b, tki, tkw, counts);
    k_offsets<<<1, 1, 0, stream>>>(counts, poff, out + (size_t)BTOK * OUT_);
    k_scatter<<<(BTOK * KSEL) / 256, 256, 0, stream>>>(tki, tkw, poff, cursors, perm, slotw);
    k_gemm1<<<dim3(ROW_TILES, HID_ / (16 * NT_)), 32, 0, stream>>>(x16, w1t, b1, perm, poff, h16, zrow);
    k_gemm2<<<dim3(ROW_TILES, OUT_ / (16 * NT_)), 32, 0, stream>>>(h16, w2t, b2, perm, slotw, poff, out);
}
